// ActorCritic_75840532513181
// MI455X (gfx1250) — compile-verified, matrix operand order checked by reading, not run on hardware
//
#include <hip/hip_runtime.h>

// ---------------------------------------------------------------------------
// CDNA5 WMMA (wave32): D(16x16 f32) = A(16x32 bf16) x B(32x16 bf16) + C
// ---------------------------------------------------------------------------
typedef __attribute__((ext_vector_type(16))) __bf16 v16bf;
typedef __attribute__((ext_vector_type(8)))  __bf16 v8bf;
typedef __attribute__((ext_vector_type(8)))  float  v8f;

#define N_UNITS     20
#define GRID_OBS    458          // 256 + 200 + 2
#define OBS_SZ      490          // GRID_OBS + 32
#define BATCH       16384

// padded channel plane: rows of 20 (left pad 2, right pad 2), 18 rows
#define RS   20                  // row stride (elements)
#define PL   360                 // plane stride = RS * 18

static __device__ __forceinline__ v8f wmma_bf16(v16bf a, v16bf b, v8f c) {
    return __builtin_amdgcn_wmma_f32_16x16x32_bf16(false, a, false, b, (short)0, c, false, false);
}

static __device__ __forceinline__ v16bf cat8(v8bf lo, v8bf hi) {
    return __builtin_shufflevector(lo, hi, 0,1,2,3,4,5,6,7,8,9,10,11,12,13,14,15);
}

// A-fragment element e -> K offset within a 32-wide k-step (half=0 base; +8 for half=1)
static __device__ __host__ constexpr int kbase(int e) { return (e < 8) ? e : (e + 8); }

// im2col LDS offset for logical k (relative to interior base address)
static __device__ __host__ constexpr int convOff(int k) {
    int cin = k / 9, r = k % 9, kh = r / 3, kw = r % 3;
    return cin * PL + (kh - 1) * RS + (kw - 1);
}

static __device__ __forceinline__ unsigned pk2(float a, float b) {
    __bf16 b0 = (__bf16)a, b1 = (__bf16)b;
    return (unsigned)__builtin_bit_cast(unsigned short, b0) |
           ((unsigned)__builtin_bit_cast(unsigned short, b1) << 16);
}

// ---------------------------------------------------------------------------
// Weight packer: w is (Nreal x Kreal) row-major f32 (reference does x @ w.T).
// Packed layout: [nt][ks][lane(32)][elem(16)] bf16, zero-padded OOB.
// ---------------------------------------------------------------------------
__global__ void pack_w(const float* __restrict__ w, int Kreal, int Nreal,
                       int KS, int NT, __bf16* __restrict__ dst) {
    int idx = blockIdx.x * 256 + threadIdx.x;
    int total = NT * KS * 512;
    if (idx >= total) return;
    int e    = idx & 15;
    int lane = (idx >> 4) & 31;
    int rest = idx >> 9;
    int ks   = rest % KS;
    int nt   = rest / KS;
    int n    = nt * 16 + (lane & 15);
    int half = lane >> 4;
    int k    = ks * 32 + kbase(e) + half * 8;
    float v  = (n < Nreal && k < Kreal) ? w[(long)n * Kreal + k] : 0.0f;
    dst[idx] = (__bf16)v;
}

__global__ void make_head_bias(const float* __restrict__ ab,
                               const float* __restrict__ cb,
                               float* __restrict__ hb) {
    int i = threadIdx.x;
    if (i < 16)       hb[i] = ab[i];
    else if (i == 16) hb[i] = cb[0];
    else if (i < 32)  hb[i] = 0.0f;
}

// ---------------------------------------------------------------------------
// One conv layer as GEMM: M=256 pixels, K=Cin*9 (KSN ksteps), N=NT*16.
// src = padded planes in LDS. Each wave: 2 m-tiles, A-frag shared across NT.
// Fully unrolled ksteps so (cin,kh,kw) offsets are compile-time constants;
// lane-half handled branchlessly via mask-add.
// ---------------------------------------------------------------------------
template<int KSN, int NT, bool POOL>
static __device__ __forceinline__ void conv_layer(
        const __bf16* __restrict__ src,
        const __bf16* __restrict__ wp, const float* __restrict__ bias,
        __bf16* __restrict__ dst, float* __restrict__ pool,
        int wave, int lane) {
    const int m = lane & 15, half = lane >> 4;
    const int hm = -half;                          // 0 or all-ones
#pragma unroll
    for (int mtl = 0; mtl < 2; ++mtl) {
        const int mt  = wave * 2 + mtl;            // output row y
        const int p20 = (mt + 1) * RS + m + 2;     // interior base for pixel (y=mt,x=m)
        v8f acc[NT];
#pragma unroll
        for (int n = 0; n < NT; ++n) acc[n] = (v8f){0,0,0,0,0,0,0,0};
#pragma unroll
        for (int ks = 0; ks < KSN; ++ks) {
            v16bf a;
#pragma unroll
            for (int e = 0; e < 16; ++e) {
                const int k0 = ks * 32 + kbase(e);
                const int d0 = convOff(k0);
                const int dd = convOff(k0 + 8) - d0;
                a[e] = src[p20 + d0 + (hm & dd)];  // unconditional ds_load_u16
            }
#pragma unroll
            for (int n = 0; n < NT; ++n) {
                v16bf bb = *(const v16bf*)(wp + ((n * KSN + ks) * 32 + lane) * 16);
                acc[n] = wmma_bf16(a, bb, acc[n]);
            }
        }
#pragma unroll
        for (int n = 0; n < NT; ++n) {
            const int ch = n * 16 + m;             // D: N = lane&15
            const float bs = bias[ch];
            if (POOL) {
                float s0 = 0.f, s1 = 0.f;
#pragma unroll
                for (int j = 0; j < 4; ++j) {
                    float v0 = acc[n][j]     + bs; s0 += (v0 > 0.f ? v0 : 0.f);
                    float v1 = acc[n][j + 4] + bs; s1 += (v1 > 0.f ? v1 : 0.f);
                }
                int base = ch * 16 + (mt >> 2) * 4 + 2 * half;   // x = 8*half + j
                atomicAdd(&pool[base],     s0 * 0.0625f);
                atomicAdd(&pool[base + 1], s1 * 0.0625f);
            } else {
                // D rows: x = 8*half + j, y = mt; consecutive -> packed b32 stores
                unsigned* d32 = (unsigned*)(dst + ch * PL + (mt + 1) * RS + 2 + 8 * half);
#pragma unroll
                for (int j = 0; j < 4; ++j) {
                    float v0 = acc[n][2*j]   + bs; v0 = v0 > 0.f ? v0 : 0.f;
                    float v1 = acc[n][2*j+1] + bs; v1 = v1 > 0.f ? v1 : 0.f;
                    d32[j] = pk2(v0, v1);
                }
            }
        }
    }
}

// ---------------------------------------------------------------------------
// Conv tower: one workgroup (256 thr = 8 waves) per sample; all intermediates
// in LDS as zero-bordered padded planes; pool fused into conv3 via ds_add_f32.
// ---------------------------------------------------------------------------
#define SG_OFF   0                     // grid: 9 planes
#define SC1_OFF  (9 * PL)              // conv1 out: 32 planes
#define SC2_OFF  (SC1_OFF + 32 * PL)   // conv2 out: 64 planes
#define SBF_TOT  (SC2_OFF + 64 * PL)   // 37800 bf16 elements
#define SMEM_BYTES (SBF_TOT * 2 + 1024 * 4)   // + pool (f32[1024]) = 79696

__global__ __launch_bounds__(256)
void conv_tower(const float* __restrict__ obs,
                const __bf16* __restrict__ w1p, const float* __restrict__ b1,
                const __bf16* __restrict__ w2p, const float* __restrict__ b2,
                const __bf16* __restrict__ w3p, const float* __restrict__ b3,
                __bf16* __restrict__ pooled) {
    __shared__ __align__(16) unsigned char smem[SMEM_BYTES];
    __bf16* sGrid = (__bf16*)smem;
    __bf16* sC1   = sGrid + SC1_OFF;
    __bf16* sC2   = sGrid + SC2_OFF;
    float*  sPool = (float*)(smem + SBF_TOT * 2);

    const int b   = blockIdx.x;
    const int tid = threadIdx.x;
    const float* o = obs + (long)b * OBS_SZ;

    // zero all planes (borders must be 0) + pool
    for (int i = tid; i < SMEM_BYTES / 16; i += 256) {
        int4 z{0, 0, 0, 0};
        ((int4*)smem)[i] = z;
    }
    __syncthreads();

    // ---- preprocess: terrain + unit scatter ----
    {
        int y = tid >> 4, x = tid & 15;
        sGrid[(y + 1) * RS + x + 2] = (__bf16)o[tid];      // channel 0
    }
    if (tid < N_UNITS) {
        const float* u = o + 256 + tid * 10;
        float team = u[1], rf = u[2], cf = u[3];
        float hp = u[4], mv = u[5], ac = u[6];
        if (hp > 0.0f) {
            int r = (int)(rf * 15.0f + 0.5f);
            int c = (int)(cf * 15.0f + 0.5f);
            int p = (r + 1) * RS + c + 2;
            bool blue = team < 0.5f;
            int chO = blue ? 1 : 2, chH = blue ? 3 : 4;
            int chM = blue ? 5 : 7, chA = blue ? 6 : 8;
            sGrid[chO * PL + p] = (__bf16)1.0f;            // cells distinct/sample
            sGrid[chH * PL + p] = (__bf16)hp;
            sGrid[chM * PL + p] = (__bf16)mv;
            sGrid[chA * PL + p] = (__bf16)ac;
        }
    }
    __syncthreads();

    const int wave = tid >> 5, lane = tid & 31;
    conv_layer<3, 2, false>(sGrid, w1p, b1, sC1, nullptr, wave, lane);  // K=81->96
    __syncthreads();
    conv_layer<9, 4, false>(sC1, w2p, b2, sC2, nullptr, wave, lane);    // K=288
    __syncthreads();
    conv_layer<18, 4, true>(sC2, w3p, b3, nullptr, sPool, wave, lane);  // K=576 +pool
    __syncthreads();

    // write pooled features (64ch x 4 x 4), 4 consecutive values per thread
    {
        float4 v = ((const float4*)sPool)[tid];
        unsigned lo = pk2(v.x, v.y), hi = pk2(v.z, v.w);
        uint2 pkt{lo, hi};
        *(uint2*)(pooled + (long)b * 1024 + tid * 4) = pkt;
    }
}

// ---------------------------------------------------------------------------
// Generic batched dense layer: one 16x16 tile per wave. A rows = batch.
// Exact-K split sources (KS0 from A0, KS1 from A1) -> unconditional vector
// loads. Optional f32 A0 (converted), relu, heads mode writing d_out.
// ---------------------------------------------------------------------------
__global__ __launch_bounds__(256)
void dense_gemm(const void* __restrict__ A0, int lda0, int KS0, int a0f32,
                const __bf16* __restrict__ A1, int lda1, int KS1,
                const __bf16* __restrict__ Wp, const float* __restrict__ bias,
                int NT,
                __bf16* __restrict__ outB, int ldo, int do_relu,
                float* __restrict__ outHeads) {
    const int gw   = blockIdx.x * 8 + (threadIdx.x >> 5);
    const int lane = threadIdx.x & 31;
    const int m = lane & 15, half = lane >> 4;
    const int nt = gw % NT, mt = gw / NT;
    const long row = (long)mt * 16 + m;
    const int KS = KS0 + KS1;

    v8f acc = {0.f, 0.f, 0.f, 0.f, 0.f, 0.f, 0.f, 0.f};
    for (int ks = 0; ks < KS0; ++ks) {
        v16bf a;
        if (a0f32) {
            const float* p = (const float*)A0 + row * lda0 + ks * 32;
#pragma unroll
            for (int e = 0; e < 8; ++e) {
                a[e]     = (__bf16)p[half * 8 + e];
                a[e + 8] = (__bf16)p[16 + half * 8 + e];
            }
        } else {
            const __bf16* p = (const __bf16*)A0 + row * lda0 + ks * 32;
            a = cat8(*(const v8bf*)(p + half * 8), *(const v8bf*)(p + 16 + half * 8));
        }
        v16bf bb = *(const v16bf*)(Wp + (((long)nt * KS + ks) * 32 + lane) * 16);
        acc = wmma_bf16(a, bb, acc);
    }
    for (int ks = 0; ks < KS1; ++ks) {
        const __bf16* p = A1 + row * lda1 + ks * 32;
        v16bf a  = cat8(*(const v8bf*)(p + half * 8), *(const v8bf*)(p + 16 + half * 8));
        v16bf bb = *(const v16bf*)(Wp + (((long)nt * KS + KS0 + ks) * 32 + lane) * 16);
        acc = wmma_bf16(a, bb, acc);
    }

    const int n = nt * 16 + m;
    const float bs = bias ? bias[n] : 0.0f;
#pragma unroll
    for (int j = 0; j < 8; ++j) {
        long r2 = (long)mt * 16 + j + 8 * half;
        float v = acc[j] + bs;
        if (outHeads) {                        // heads: N padded to 32
            if (n < 16)       outHeads[r2 * 16 + n] = v;            // logits
            else if (n == 16) outHeads[(long)BATCH * 16 + r2] = v;  // value
        } else {
            if (do_relu) v = v > 0.f ? v : 0.f;
            outB[r2 * ldo + n] = (__bf16)v;
        }
    }
}

// ---------------------------------------------------------------------------
// Host-side launcher
// ---------------------------------------------------------------------------
static inline size_t align_up(size_t x, size_t a) { return (x + a - 1) & ~(a - 1); }

extern "C" void kernel_launch(void* const* d_in, const int* in_sizes, int n_in,
                              void* d_out, int out_size, void* d_ws, size_t ws_size,
                              hipStream_t stream) {
    const float* obs = (const float*)d_in[0];
    const float* c1w = (const float*)d_in[1];
    const float* c1b = (const float*)d_in[2];
    const float* c2w = (const float*)d_in[3];
    const float* c2b = (const float*)d_in[4];
    const float* c3w = (const float*)d_in[5];
    const float* c3b = (const float*)d_in[6];
    const float* fcw = (const float*)d_in[7];
    const float* fcb = (const float*)d_in[8];
    const float* u1w = (const float*)d_in[9];
    const float* u1b = (const float*)d_in[10];
    const float* u2w = (const float*)d_in[11];
    const float* u2b = (const float*)d_in[12];
    const float* t1w = (const float*)d_in[13];
    const float* t1b = (const float*)d_in[14];
    const float* t2w = (const float*)d_in[15];
    const float* t2b = (const float*)d_in[16];
    const float* aw  = (const float*)d_in[17];
    const float* ab  = (const float*)d_in[18];
    const float* cw  = (const float*)d_in[19];
    const float* cb  = (const float*)d_in[20];

    uint8_t* ws = (uint8_t*)d_ws;
    size_t cur = 0;
    auto take = [&](size_t bytes) -> uint8_t* {
        uint8_t* p = ws + cur;
        cur = align_up(cur + bytes, 256);
        return p;
    };

    // packed weights [nt][ks][32][16] bf16
    __bf16* w1p = (__bf16*)take((size_t)2  * 3  * 512 * 2);
    __bf16* w2p = (__bf16*)take((size_t)4  * 9  * 512 * 2);
    __bf16* w3p = (__bf16*)take((size_t)4  * 18 * 512 * 2);
    __bf16* fcp = (__bf16*)take((size_t)16 * 32 * 512 * 2);
    __bf16* u1p = (__bf16*)take((size_t)4  * 1  * 512 * 2);
    __bf16* u2p = (__bf16*)take((size_t)4  * 2  * 512 * 2);
    __bf16* t1p = (__bf16*)take((size_t)16 * 10 * 512 * 2);
    __bf16* t2p = (__bf16*)take((size_t)16 * 8  * 512 * 2);
    __bf16* hdp = (__bf16*)take((size_t)2  * 8  * 512 * 2);
    float*  hdb = (float*) take(32 * sizeof(float));

    // activations (bf16)
    __bf16* pooled = (__bf16*)take((size_t)BATCH * 1024 * 2);
    __bf16* cnn    = (__bf16*)take((size_t)BATCH * 256 * 2);
    __bf16* uh     = (__bf16*)take((size_t)BATCH * 64 * 2);
    __bf16* uf     = (__bf16*)take((size_t)BATCH * 64 * 2);
    __bf16* t1o    = (__bf16*)take((size_t)BATCH * 256 * 2);
    __bf16* t2o    = (__bf16*)take((size_t)BATCH * 256 * 2);

    auto packN = [&](const float* w, int Kreal, int Nreal, int KS, int NT, __bf16* dst) {
        int total = NT * KS * 512;
        pack_w<<<(total + 255) / 256, 256, 0, stream>>>(w, Kreal, Nreal, KS, NT, dst);
    };
    // conv weights (O, I, 3, 3) contiguous == (O x I*9) row-major
    packN(c1w, 81,   32,  3,  2,  w1p);
    packN(c2w, 288,  64,  9,  4,  w2p);
    packN(c3w, 576,  64,  18, 4,  w3p);
    packN(fcw, 1024, 256, 32, 16, fcp);
    packN(u1w, 32,   64,  1,  4,  u1p);
    packN(u2w, 64,   64,  2,  4,  u2p);
    packN(t1w, 320,  256, 10, 16, t1p);
    packN(t2w, 256,  256, 8,  16, t2p);
    packN(aw,  256,  16,  8,  1,  hdp);            // nt=0: actor
    packN(cw,  256,  1,   8,  1,  hdp + 8 * 512);  // nt=1: critic
    make_head_bias<<<1, 32, 0, stream>>>(ab, cb, hdb);

    // conv tower (preprocess + 3 convs + fused pool), one WG per sample
    conv_tower<<<BATCH, 256, 0, stream>>>(obs, w1p, c1b, w2p, c2b, w3p, c3b, pooled);

    const int MT = BATCH / 16;  // 1024 row-tiles
    auto dense = [&](const void* A0, int lda0, int KS0, int a0f,
                     const __bf16* A1, int lda1, int KS1,
                     const __bf16* Wp, const float* bias, int NT,
                     __bf16* outB, int ldo, int relu, float* heads) {
        int waves = MT * NT;
        dense_gemm<<<waves / 8, 256, 0, stream>>>(A0, lda0, KS0, a0f, A1, lda1, KS1,
                                                  Wp, bias, NT, outB, ldo, relu, heads);
    };

    // unit MLP (independent of conv tower)
    dense(obs + GRID_OBS, OBS_SZ, 1, 1, nullptr, 0, 0, u1p, u1b, 4,  uh,  64,  1, nullptr);
    dense(uh,  64,  2, 0, nullptr, 0, 0,               u2p, u2b, 4,  uf,  64,  1, nullptr);
    // fc on pooled conv features
    dense(pooled, 1024, 32, 0, nullptr, 0, 0,          fcp, fcb, 16, cnn, 256, 1, nullptr);
    // trunk: concat(cnn 256, unit 64) via dual A sources
    dense(cnn, 256, 8, 0, uf, 64, 2,                   t1p, t1b, 16, t1o, 256, 1, nullptr);
    dense(t1o, 256, 8, 0, nullptr, 0, 0,               t2p, t2b, 16, t2o, 256, 1, nullptr);
    // heads: actor(16) + critic(1) padded to N=32, writes d_out directly
    dense(t2o, 256, 8, 0, nullptr, 0, 0,               hdp, hdb, 2,  nullptr, 0, 0, (float*)d_out);
}
